// LSTM_27152783245909
// MI455X (gfx1250) — matrix operand order repaired and checked by reading, not compiled, not run on hardware
//
#include <hip/hip_runtime.h>
#include <math.h>

#define T_STEPS 256
#define BATCH   64
#define IN_D    512
#define HID     1024
#define OUT_D   512
#define GATES   (4 * HID)      // 4096
#define NWG_REC 128

typedef __attribute__((ext_vector_type(16))) __bf16 v16bf;
typedef __attribute__((ext_vector_type(8)))  __bf16 v8bf;
typedef __attribute__((ext_vector_type(8)))  float  v8f;

// ---------------- WMMA fragment helpers (CDNA5 16x16x32 bf16 layouts) ----------------
// A (16x32, MxK): lane L -> row = L%16; vector elems e = K (L/16)*16 + e
static __device__ __forceinline__ v16bf load_a_frag(const __bf16* base, int ld,
                                                    int rowBase, int kBase, int lane) {
  const int r  = rowBase + (lane & 15);
  const int kg = (lane >> 4) << 4;
  const __bf16* p = base + (size_t)r * ld + kBase + kg;
  v8bf lo = *(const v8bf*)(p);        // K = kBase+kg .. +7
  v8bf hi = *(const v8bf*)(p + 8);    // K = kBase+kg+8 .. +15
  return __builtin_shufflevector(lo, hi, 0,1,2,3,4,5,6,7,8,9,10,11,12,13,14,15);
}

// B (32x16, KxN): lane L -> col = L%16; elems e = K (L/16)*16 + e.
// W is row-major [N_total, ld] with K contiguous -> one 32B load per lane.
static __device__ __forceinline__ v16bf load_b_frag(const __bf16* W, int ld,
                                                    int nBase, int kBase, int lane) {
  const int n = nBase + (lane & 15);
  const __bf16* p = W + (size_t)n * ld + kBase + ((lane >> 4) << 4);
  return *(const v16bf*)(p);
}

static __device__ __forceinline__ v8f wmma_bf16(v16bf a, v16bf b, v8f c) {
  return __builtin_amdgcn_wmma_f32_16x16x32_bf16(false, a, false, b, (short)0, c, false, false);
}

static __device__ __forceinline__ float sigmoidf_fast(float x) {
  return 1.0f / (1.0f + __expf(-x));
}

// ---------------- Prep: convert weights/x to bf16, init h buffer, zero barrier ----------------
__global__ __launch_bounds__(256)
void lstm_prep(const float* __restrict__ W_ih, const float* __restrict__ W_hh,
               const float* __restrict__ W_out, const float* __restrict__ x,
               const float* __restrict__ h0,
               __bf16* __restrict__ Wih_b, __bf16* __restrict__ Whh_b,
               __bf16* __restrict__ Wout_b, __bf16* __restrict__ xs_b,
               __bf16* __restrict__ hbuf, unsigned* __restrict__ bar) {
  const size_t N0 = (size_t)GATES * IN_D;             // W_ih
  const size_t N1 = N0 + (size_t)GATES * HID;         // W_hh
  const size_t N2 = N1 + (size_t)OUT_D * HID;         // W_out
  const size_t N3 = N2 + (size_t)T_STEPS * BATCH * IN_D;  // xs (transposed)
  const size_t N4 = N3 + (size_t)BATCH * HID;         // h init
  const size_t N5 = N4 + 2;                           // barrier words
  const size_t stride = (size_t)gridDim.x * blockDim.x;
  for (size_t e = (size_t)blockIdx.x * blockDim.x + threadIdx.x; e < N5; e += stride) {
    if (e < N0) {
      Wih_b[e] = (__bf16)W_ih[e];
    } else if (e < N1) {
      size_t i = e - N0; Whh_b[i] = (__bf16)W_hh[i];
    } else if (e < N2) {
      size_t i = e - N1; Wout_b[i] = (__bf16)W_out[i];
    } else if (e < N3) {
      size_t i = e - N2;
      size_t row = i >> 9;            // t*64 + b
      size_t ii  = i & 511;
      size_t t   = row >> 6;
      size_t b   = row & 63;
      // x layout [B, I, T]
      xs_b[i] = (__bf16)x[(b * IN_D + ii) * T_STEPS + t];
    } else if (e < N4) {
      size_t i = e - N3;
      hbuf[i] = (__bf16)h0[i & (HID - 1)];  // broadcast h0 over batch
    } else {
      bar[e - N4] = 0u;
    }
  }
}

// ---------------- Grid-wide barrier (monotonic counters, init'd by prep) ----------------
static __device__ __forceinline__ void grid_barrier(unsigned* bar, unsigned step) {
  __syncthreads();
  if (threadIdx.x == 0) {
    __threadfence();
    unsigned prev = atomicAdd(&bar[0], 1u);
    if (prev == (unsigned)NWG_REC * (step + 1u) - 1u) {
      atomicAdd(&bar[1], 1u);           // release this generation
    }
    while (atomicAdd(&bar[1], 0u) < step + 1u) {
      __builtin_amdgcn_s_sleep(2);
    }
    __threadfence();
  }
  __syncthreads();
}

// ---------------- Persistent recurrent kernel ----------------
// Grid: 128 WGs x 256 thr (8 waves). WG = (hidden slice of 16 cols) x (32 batch rows).
// Wave w: gate g = w%4, mtile = mpair*2 + w/4  -> one 16x16 WMMA tile, K = 512(x) + 1024(h).
__global__ __launch_bounds__(256)
void lstm_recurrent(const __bf16* __restrict__ xs,     // [T*B, 512] bf16
                    const __bf16* __restrict__ Wih,    // [4096, 512] bf16
                    const __bf16* __restrict__ Whh,    // [4096, 1024] bf16
                    const float* __restrict__ b_ih, const float* __restrict__ b_hh,
                    const float* __restrict__ c0,
                    __bf16* __restrict__ hbuf,         // [2][64][1024] bf16 ping-pong
                    __bf16* __restrict__ c_all,        // [T*64][1024] bf16
                    unsigned* __restrict__ bar) {
  __shared__ float gates[4][32][16];
  const int tid   = threadIdx.x;
  const int lane  = tid & 31;
  const int wid   = tid >> 5;
  const int slice = blockIdx.x & 63;     // hidden slice id
  const int mpair = blockIdx.x >> 6;     // 0..1 -> batch rows mpair*32..+31
  const int nHid  = slice << 4;          // hidden col base
  const int gate  = wid & 3;
  const int mtile = (mpair << 1) + (wid >> 2);
  const int rowBase = mtile << 4;        // batch row base of this wave's tile
  const int gcol  = gate * HID + nHid;   // column base in [0,4096)

  const int nlane = lane & 15;
  const float bias = b_ih[gcol + nlane] + b_hh[gcol + nlane];

  // Per-thread cell state: 2 elements of the WG's [32 x 16] slice.
  float creg[2];
#pragma unroll
  for (int j = 0; j < 2; ++j) {
    int idx = (tid << 1) + j;
    creg[j] = c0[nHid + (idx & 15)];
  }

  for (int t = 0; t < T_STEPS; ++t) {
    const __bf16* hcur = hbuf + (size_t)(t & 1) * (BATCH * HID);
    const __bf16* xrow = xs + (size_t)t * BATCH * IN_D;

    v8f acc = {};
    // x contribution: K = 512
#pragma unroll 4
    for (int kt = 0; kt < IN_D; kt += 32) {
      v16bf a = load_a_frag(xrow, IN_D, rowBase, kt, lane);
      v16bf b = load_b_frag(Wih, IN_D, gcol, kt, lane);
      acc = wmma_bf16(a, b, acc);
    }
    // h contribution: K = 1024
#pragma unroll 4
    for (int kt = 0; kt < HID; kt += 32) {
      v16bf a = load_a_frag(hcur, HID, rowBase, kt, lane);
      v16bf b = load_b_frag(Whh, HID, gcol, kt, lane);
      acc = wmma_bf16(a, b, acc);
    }

    // Dump gate tile to LDS: acc elem r -> (local row, col)
    const int lr0 = ((wid >> 2) << 4) + ((lane >> 4) << 3);
#pragma unroll
    for (int r = 0; r < 8; ++r)
      gates[gate][lr0 + r][nlane] = acc[r] + bias;
    __syncthreads();

    // Elementwise LSTM cell update for this WG's [32 x 16] slice.
#pragma unroll
    for (int j = 0; j < 2; ++j) {
      int idx = (tid << 1) + j;
      int br  = idx >> 4;               // local batch row 0..31
      int nn  = idx & 15;
      float ig = sigmoidf_fast(gates[0][br][nn]);
      float fg = sigmoidf_fast(gates[1][br][nn]);
      float gg = tanhf(gates[2][br][nn]);
      float og = sigmoidf_fast(gates[3][br][nn]);
      float cn = fg * creg[j] + ig * gg;
      float hn = og * tanhf(cn);
      creg[j] = cn;
      int gb = (mpair << 5) + br;       // global batch row
      int gh = nHid + nn;               // global hidden col
      hbuf[(size_t)((t + 1) & 1) * (BATCH * HID) + (size_t)gb * HID + gh] = (__bf16)hn;
      c_all[((size_t)t * BATCH + gb) * HID + gh] = (__bf16)cn;
    }
    grid_barrier(bar, (unsigned)t);     // also protects gates-LDS reuse
  }
}

// ---------------- Output head: logits = c_all @ W_out^T + b_out; row softmax ----------------
// Grid: 1024 WGs; WG = one 16-row tile x all 512 output cols. 8 waves x 4 N-tiles.
__global__ __launch_bounds__(256)
void lstm_head(const __bf16* __restrict__ c_all, const __bf16* __restrict__ Wout,
               const float* __restrict__ b_out, float* __restrict__ out) {
  __shared__ float logits[16][OUT_D];
  __shared__ float red[16][16];
  const int tid  = threadIdx.x;
  const int lane = tid & 31;
  const int wid  = tid >> 5;
  const int rowBase = blockIdx.x << 4;   // 0..16368

  v8f acc[4] = {};
#pragma unroll 4
  for (int kt = 0; kt < HID; kt += 32) {
    v16bf a = load_a_frag(c_all, HID, rowBase, kt, lane);
#pragma unroll
    for (int q = 0; q < 4; ++q) {
      v16bf b = load_b_frag(Wout, HID, ((wid << 2) + q) << 4, kt, lane);
      acc[q] = wmma_bf16(a, b, acc[q]);
    }
  }
  const int r0 = (lane >> 4) << 3;
#pragma unroll
  for (int q = 0; q < 4; ++q) {
    int col = (((wid << 2) + q) << 4) + (lane & 15);
    float bo = b_out[col];
#pragma unroll
    for (int r = 0; r < 8; ++r)
      logits[r0 + r][col] = acc[q][r] + bo;
  }
  __syncthreads();

  // Softmax: 16 threads per row, 32 cols per thread.
  const int row = tid >> 4;
  const int j   = tid & 15;
  float mx = -3.4e38f;
#pragma unroll 4
  for (int k = 0; k < 32; ++k) mx = fmaxf(mx, logits[row][(j << 5) + k]);
  red[row][j] = mx;
  __syncthreads();
  float rmax = red[row][0];
#pragma unroll
  for (int k = 1; k < 16; ++k) rmax = fmaxf(rmax, red[row][k]);
  __syncthreads();
  float s = 0.0f;
#pragma unroll 4
  for (int k = 0; k < 32; ++k) {
    float e = __expf(logits[row][(j << 5) + k] - rmax);
    logits[row][(j << 5) + k] = e;
    s += e;
  }
  red[row][j] = s;
  __syncthreads();
  float rsum = 0.0f;
#pragma unroll
  for (int k = 0; k < 16; ++k) rsum += red[row][k];
  const float inv = 1.0f / rsum;
#pragma unroll 4
  for (int k = 0; k < 32; ++k)
    out[((size_t)(rowBase + row)) * OUT_D + (j << 5) + k] = logits[row][(j << 5) + k] * inv;
}

// ---------------- Host launcher ----------------
extern "C" void kernel_launch(void* const* d_in, const int* in_sizes, int n_in,
                              void* d_out, int out_size, void* d_ws, size_t ws_size,
                              hipStream_t stream) {
  const float* x     = (const float*)d_in[0];
  const float* W_ih  = (const float*)d_in[1];
  const float* W_hh  = (const float*)d_in[2];
  const float* b_ih  = (const float*)d_in[3];
  const float* b_hh  = (const float*)d_in[4];
  const float* W_out = (const float*)d_in[5];
  const float* b_out = (const float*)d_in[6];
  const float* h0    = (const float*)d_in[7];
  const float* c0    = (const float*)d_in[8];
  float* out = (float*)d_out;
  (void)in_sizes; (void)n_in; (void)out_size; (void)h0;

  char* ws = (char*)d_ws;
  size_t off = 0;
  __bf16* Wih_b  = (__bf16*)(ws + off); off += (size_t)GATES * IN_D * 2;
  __bf16* Whh_b  = (__bf16*)(ws + off); off += (size_t)GATES * HID * 2;
  __bf16* Wout_b = (__bf16*)(ws + off); off += (size_t)OUT_D * HID * 2;
  __bf16* xs_b   = (__bf16*)(ws + off); off += (size_t)T_STEPS * BATCH * IN_D * 2;
  __bf16* hbuf   = (__bf16*)(ws + off); off += (size_t)2 * BATCH * HID * 2;
  __bf16* c_all  = (__bf16*)(ws + off); off += (size_t)T_STEPS * BATCH * HID * 2;
  unsigned* bar  = (unsigned*)(ws + off); off += 256;
  if (ws_size < off) return;  // workspace too small: refuse rather than corrupt

  lstm_prep<<<4096, 256, 0, stream>>>(W_ih, W_hh, W_out, x, h0,
                                      Wih_b, Whh_b, Wout_b, xs_b, hbuf, bar);
  lstm_recurrent<<<NWG_REC, 256, 0, stream>>>(xs_b, Wih_b, Whh_b, b_ih, b_hh, c0,
                                              hbuf, c_all, bar);
  lstm_head<<<(T_STEPS * BATCH) / 16, 256, 0, stream>>>(c_all, Wout_b, b_out, out);
}
